// MultiHeadSelfAttention_90615220011017
// MI455X (gfx1250) — matrix operand order repaired and checked by reading, not compiled, not run on hardware
//
#include <hip/hip_runtime.h>
#include <stdint.h>
#include <stddef.h>

// ---------------------------------------------------------------------------
// Problem constants (from reference): B=2, S=2048, D_MODEL=1024, H=16, HD=64
// ---------------------------------------------------------------------------
#define DM   1024
#define BB   2
#define SS   2048
#define HH   16
#define HD   64
#define MTOT (BB * SS)   // 4096 rows

typedef __attribute__((ext_vector_type(16))) __bf16       v16bf;
typedef __attribute__((ext_vector_type(8)))  float        v8f;
typedef __attribute__((ext_vector_type(4)))  unsigned int u32x4;

union ABFrag { u32x4 u[2]; v16bf v; };

// ds_swizzle group-of-32 XOR shuffle: offset = (xor_mask<<10) | and_mask(0x1f)
#define SWZ_XOR_F(x, m) \
  __int_as_float(__builtin_amdgcn_ds_swizzle(__float_as_int(x), (((m) << 10) | 0x1f)))

// CDNA5 async copy: global memory -> LDS, tracked by ASYNCcnt.
// lds_off is the 32-bit LDS byte offset (low 32 bits of the generic address).
#define ASYNC_G2L_B128(lds_off, gptr)                                \
  asm volatile("global_load_async_to_lds_b128 %0, %1, off"           \
               :: "v"(lds_off), "v"(gptr) : "memory")

#define WAIT_ASYNC_0() asm volatile("s_wait_asynccnt 0x0" ::: "memory")

// ---------------------------------------------------------------------------
// fp32 -> bf16 conversion
// ---------------------------------------------------------------------------
__global__ void f32_to_bf16_kernel(const float* __restrict__ src,
                                   __bf16* __restrict__ dst, int n) {
  int i = blockIdx.x * blockDim.x + threadIdx.x;
  if (i < n) dst[i] = (__bf16)src[i];
}

// ---------------------------------------------------------------------------
// GEMM: out[m,n] = sum_k A[m,k] * W[n,k]   (x @ W^T, nn.Linear semantics)
// M=4096, N=1024, K=1024. Block = 4 waves, block tile 128x128, K-step 32.
// Double-buffered LDS staging via global_load_async_to_lds_b128 (ASYNCcnt).
// Tile rows padded 64B->80B so 16-lane ds_load_b128 fragment reads are
// spread across distinct bank groups.
// MODE 0: store bf16 scatter to [b,h,s,d]        (Q, K)
// MODE 1: store bf16 scatter to [b,h,d,s]        (V transposed for PV GEMM)
// MODE 2: store f32 row-major + bias             (output projection)
// ---------------------------------------------------------------------------
#define ROWPAD 40   // 32 payload elems + 8 pad elems (80 bytes)

template <int MODE>
__global__ __launch_bounds__(128) void gemm_xwt_kernel(
    const __bf16* __restrict__ A, const __bf16* __restrict__ W,
    void* __restrict__ out, const float* __restrict__ bias) {
  __shared__ __align__(16) __bf16 Atile[2][128][ROWPAD];  // 20 KB
  __shared__ __align__(16) __bf16 Btile[2][128][ROWPAD];  // 20 KB

  const int tid  = threadIdx.x;
  const int lane = tid & 31;
  const int wave = tid >> 5;
  const int l15  = lane & 15;
  const int hi   = lane >> 4;                       // lane half: 0 or 1
  const int tile_m = blockIdx.y * 128;
  const int tile_n = blockIdx.x * 128;
  const int wm = (wave >> 1) * 64;                  // wave sub-tile within block
  const int wn = (wave & 1) * 64;

  // Issue async loads for one 128x32 A tile + one 128x32 B tile into buf.
  // 512 16-byte chunks per tile; 128 threads x 4 chunks each.
  auto issue_tiles = [&](int buf, int k0) {
#pragma unroll
    for (int p2 = 0; p2 < 4; ++p2) {
      const int c   = tid + p2 * 128;
      const int row = c >> 2;
      const int cc  = (c & 3) * 8;                  // element column (x8 bf16 = 16B)
      const unsigned la = (unsigned)(uintptr_t)&Atile[buf][row][cc];
      const __bf16* ga  = A + (size_t)(tile_m + row) * DM + k0 + cc;
      ASYNC_G2L_B128(la, ga);
      const unsigned lb = (unsigned)(uintptr_t)&Btile[buf][row][cc];
      const __bf16* gb  = W + (size_t)(tile_n + row) * DM + k0 + cc;
      ASYNC_G2L_B128(lb, gb);
    }
  };

  v8f acc[4][4];
#pragma unroll
  for (int mi = 0; mi < 4; ++mi)
#pragma unroll
    for (int ni = 0; ni < 4; ++ni) acc[mi][ni] = v8f{};

  issue_tiles(0, 0);
  int buf = 0;
  for (int t = 0; t < DM / 32; ++t) {
    WAIT_ASYNC_0();        // own async loads for buf have landed in LDS
    __syncthreads();       // => all waves' loads landed; prev buffer consumers done
    if (t + 1 < DM / 32) issue_tiles(buf ^ 1, (t + 1) * 32);

    ABFrag a[4], b[4];
#pragma unroll
    for (int mi = 0; mi < 4; ++mi) {
      const __bf16* p = &Atile[buf][wm + mi * 16 + l15][hi * 8];
      a[mi].u[0] = *(const u32x4*)p;                // ds_load_b128
      a[mi].u[1] = *(const u32x4*)(p + 16);
    }
#pragma unroll
    for (int ni = 0; ni < 4; ++ni) {
      const __bf16* p = &Btile[buf][wn + ni * 16 + l15][hi * 8];
      b[ni].u[0] = *(const u32x4*)p;
      b[ni].u[1] = *(const u32x4*)(p + 16);
    }
#pragma unroll
    for (int mi = 0; mi < 4; ++mi)
#pragma unroll
      for (int ni = 0; ni < 4; ++ni)
        acc[mi][ni] = __builtin_amdgcn_wmma_f32_16x16x32_bf16(
            false, a[mi].v, false, b[ni].v, (short)0, acc[mi][ni], false, false);
    buf ^= 1;
  }

  // Epilogue. C/D layout: VGPR r -> row (r + hi*8), col = l15 within 16x16 tile.
#pragma unroll
  for (int mi = 0; mi < 4; ++mi) {
#pragma unroll
    for (int ni = 0; ni < 4; ++ni) {
      const int ncol = tile_n + wn + ni * 16 + l15;
      float bv = 0.0f;
      if constexpr (MODE == 2) bv = bias[ncol];
#pragma unroll
      for (int r = 0; r < 8; ++r) {
        const int mrow = tile_m + wm + mi * 16 + r + hi * 8;
        const float val = acc[mi][ni][r];
        if constexpr (MODE == 2) {
          ((float*)out)[(size_t)mrow * DM + ncol] = val + bv;
        } else {
          const int b_ = mrow >> 11, s = mrow & (SS - 1);
          const int h  = ncol >> 6,  d = ncol & (HD - 1);
          if constexpr (MODE == 0)
            ((__bf16*)out)[(((size_t)(b_ * HH + h)) * SS + s) * HD + d] = (__bf16)val;
          else
            ((__bf16*)out)[(((size_t)(b_ * HH + h)) * HD + d) * SS + s] = (__bf16)val;
        }
      }
    }
  }
}

// ---------------------------------------------------------------------------
// Causal flash attention. One wave owns 16 query rows; streams 32-wide key
// blocks. q/k: [b,h,s,d] bf16. vt: [b,h,d,s] bf16. ctx out: [b,s,h*64+d] bf16.
// Softmax in exp2 domain; scale folded: (1/sqrt(64)) * log2(e).
// ---------------------------------------------------------------------------
__global__ __launch_bounds__(128) void attn_kernel(
    const __bf16* __restrict__ q, const __bf16* __restrict__ k,
    const __bf16* __restrict__ vt, __bf16* __restrict__ ctx) {
  __shared__ __align__(16) __bf16 plds[4][16][32];   // P tile staging, per wave

  const int lane = threadIdx.x & 31;
  const int wave = threadIdx.x >> 5;
  const int l15  = lane & 15;
  const int hi   = lane >> 4;
  const int bh   = blockIdx.y;       // b*16 + h
  const int b_   = bh >> 4;
  const int h    = bh & 15;
  const int i0   = blockIdx.x * 64 + wave * 16;

  const __bf16* qb = q  + (size_t)bh * SS * HD;
  const __bf16* kp = k  + (size_t)bh * SS * HD;
  const __bf16* vb = vt + (size_t)bh * HD * SS;

  // Q A-fragments for d = 0..31 and 32..63
  ABFrag qa[2];
#pragma unroll
  for (int kk = 0; kk < 2; ++kk) {
    const __bf16* p = qb + (size_t)(i0 + l15) * HD + kk * 32 + hi * 8;
    qa[kk].u[0] = *(const u32x4*)p;
    qa[kk].u[1] = *(const u32x4*)(p + 16);
  }

  v8f ctxacc[4];
#pragma unroll
  for (int ni = 0; ni < 4; ++ni) ctxacc[ni] = v8f{};
  float mrow[8], lrow[8];
#pragma unroll
  for (int r = 0; r < 8; ++r) { mrow[r] = -3.0e38f; lrow[r] = 0.0f; }

  const float SC = 0.18033688011112042f;  // (1/8) * log2(e)
  const int nblk = ((i0 + 15) >> 5) + 1;  // causal: only blocks intersecting j <= i

  for (int blk = 0; blk < nblk; ++blk) {
    const int j0 = blk * 32;

    // S = Q @ K^T : two 16x16 score tiles covering 32 key columns
    v8f s[2];
#pragma unroll
    for (int jt = 0; jt < 2; ++jt) {
      v8f c = v8f{};
#pragma unroll
      for (int kk = 0; kk < 2; ++kk) {
        ABFrag kf;
        const __bf16* p = kp + (size_t)(j0 + jt * 16 + l15) * HD + kk * 32 + hi * 8;
        kf.u[0] = *(const u32x4*)p;
        kf.u[1] = *(const u32x4*)(p + 16);
        c = __builtin_amdgcn_wmma_f32_16x16x32_bf16(
            false, qa[kk].v, false, kf.v, (short)0, c, false, false);
      }
      s[jt] = c;
    }

    // scale + causal mask + per-row block max (XOR reduce across 16-lane row group)
    float rmax[8];
#pragma unroll
    for (int r = 0; r < 8; ++r) {
      const int i  = i0 + r + hi * 8;
      const int jA = j0 + l15;
      const int jB = j0 + 16 + l15;
      float a0 = s[0][r] * SC; if (jA > i) a0 = -3.0e38f;
      float a1 = s[1][r] * SC; if (jB > i) a1 = -3.0e38f;
      s[0][r] = a0; s[1][r] = a1;
      float t = fmaxf(a0, a1);
      t = fmaxf(t, SWZ_XOR_F(t, 1));
      t = fmaxf(t, SWZ_XOR_F(t, 2));
      t = fmaxf(t, SWZ_XOR_F(t, 4));
      t = fmaxf(t, SWZ_XOR_F(t, 8));
      rmax[r] = t;
    }

    // online softmax update
    float alpha[8];
#pragma unroll
    for (int r = 0; r < 8; ++r) {
      const float mn = fmaxf(mrow[r], rmax[r]);
      alpha[r] = __builtin_amdgcn_exp2f(mrow[r] - mn);
      mrow[r] = mn;
      const float p0 = __builtin_amdgcn_exp2f(s[0][r] - mn);
      const float p1 = __builtin_amdgcn_exp2f(s[1][r] - mn);
      s[0][r] = p0; s[1][r] = p1;
      float t = p0 + p1;
      t += SWZ_XOR_F(t, 1);
      t += SWZ_XOR_F(t, 2);
      t += SWZ_XOR_F(t, 4);
      t += SWZ_XOR_F(t, 8);
      lrow[r] = lrow[r] * alpha[r] + t;
    }
#pragma unroll
    for (int ni = 0; ni < 4; ++ni)
#pragma unroll
      for (int r = 0; r < 8; ++r) ctxacc[ni][r] *= alpha[r];

    // C-layout -> A-fragment layout for P via LDS round trip (same-wave DS is in-order)
#pragma unroll
    for (int r = 0; r < 8; ++r) {
      const int row = r + hi * 8;
      plds[wave][row][l15]      = (__bf16)s[0][r];
      plds[wave][row][16 + l15] = (__bf16)s[1][r];
    }
    ABFrag pf;
    {
      const __bf16* p = &plds[wave][l15][hi * 8];
      pf.u[0] = *(const u32x4*)p;
      pf.u[1] = *(const u32x4*)(p + 16);
    }

    // ctx += P @ V  (vt is [d, s]: B-operand rows are contiguous in s)
#pragma unroll
    for (int ni = 0; ni < 4; ++ni) {
      ABFrag vf;
      const __bf16* p = vb + (size_t)(ni * 16 + l15) * SS + j0 + hi * 8;
      vf.u[0] = *(const u32x4*)p;
      vf.u[1] = *(const u32x4*)(p + 16);
      ctxacc[ni] = __builtin_amdgcn_wmma_f32_16x16x32_bf16(
          false, pf.v, false, vf.v, (short)0, ctxacc[ni], false, false);
    }
  }

  // normalize + store ctx as [b, s, h*64+d] bf16 (A-operand layout for out-proj)
#pragma unroll
  for (int r = 0; r < 8; ++r) {
    const float inv = __builtin_amdgcn_rcpf(lrow[r]);
    const int i = i0 + r + hi * 8;
#pragma unroll
    for (int ni = 0; ni < 4; ++ni) {
      const int d = ni * 16 + l15;
      ctx[((size_t)(b_ * SS + i)) * DM + h * HD + d] = (__bf16)(ctxacc[ni][r] * inv);
    }
  }
}

// ---------------------------------------------------------------------------
// Host-side orchestration
// ---------------------------------------------------------------------------
extern "C" void kernel_launch(void* const* d_in, const int* in_sizes, int n_in,
                              void* d_out, int out_size, void* d_ws, size_t ws_size,
                              hipStream_t stream) {
  (void)in_sizes; (void)n_in; (void)out_size; (void)ws_size;
  const float* x  = (const float*)d_in[0];
  const float* Wq = (const float*)d_in[1];
  const float* Wk = (const float*)d_in[2];
  const float* Wv = (const float*)d_in[3];
  const float* Wo = (const float*)d_in[4];
  const float* bo = (const float*)d_in[5];

  char* ws = (char*)d_ws;
  __bf16* x_bf  = (__bf16*)ws; ws += (size_t)MTOT * DM * 2;   // 8 MB
  __bf16* wq_bf = (__bf16*)ws; ws += (size_t)DM * DM * 2;     // 2 MB
  __bf16* wk_bf = (__bf16*)ws; ws += (size_t)DM * DM * 2;
  __bf16* wv_bf = (__bf16*)ws; ws += (size_t)DM * DM * 2;
  __bf16* wo_bf = (__bf16*)ws; ws += (size_t)DM * DM * 2;
  __bf16* q_bf  = (__bf16*)ws; ws += (size_t)MTOT * DM * 2;   // [b,h,s,d]
  __bf16* k_bf  = (__bf16*)ws; ws += (size_t)MTOT * DM * 2;   // [b,h,s,d]
  __bf16* vt_bf = (__bf16*)ws; ws += (size_t)MTOT * DM * 2;   // [b,h,d,s]
  __bf16* c_bf  = (__bf16*)ws; ws += (size_t)MTOT * DM * 2;   // [b,s,h*d]

  const int nX = MTOT * DM, nW = DM * DM;
  f32_to_bf16_kernel<<<(nX + 255) / 256, 256, 0, stream>>>(x,  x_bf,  nX);
  f32_to_bf16_kernel<<<(nW + 255) / 256, 256, 0, stream>>>(Wq, wq_bf, nW);
  f32_to_bf16_kernel<<<(nW + 255) / 256, 256, 0, stream>>>(Wk, wk_bf, nW);
  f32_to_bf16_kernel<<<(nW + 255) / 256, 256, 0, stream>>>(Wv, wv_bf, nW);
  f32_to_bf16_kernel<<<(nW + 255) / 256, 256, 0, stream>>>(Wo, wo_bf, nW);

  dim3 ggrid(DM / 128, MTOT / 128);     // (8, 32)
  gemm_xwt_kernel<0><<<ggrid, 128, 0, stream>>>(x_bf, wq_bf, (void*)q_bf,  nullptr);
  gemm_xwt_kernel<0><<<ggrid, 128, 0, stream>>>(x_bf, wk_bf, (void*)k_bf,  nullptr);
  gemm_xwt_kernel<1><<<ggrid, 128, 0, stream>>>(x_bf, wv_bf, (void*)vt_bf, nullptr);

  dim3 agrid(SS / 64, BB * HH);         // (32, 32)
  attn_kernel<<<agrid, 128, 0, stream>>>(q_bf, k_bf, vt_bf, c_bf);

  gemm_xwt_kernel<2><<<ggrid, 128, 0, stream>>>(c_bf, wo_bf, d_out, bo);
}
